// MaskedSelfAttention_89455578841429
// MI455X (gfx1250) — matrix operand order repaired and checked by reading, not compiled, not run on hardware
//
#include <hip/hip_runtime.h>
#include <stdint.h>

// Problem constants (match reference)
#define E_DIM 2048
#define DK    2048
#define DV    2048
#define BATCH 4
#define SEQ   4096
#define SM_SCALE 0.02209708691207961f   // 1/sqrt(2048)

typedef __attribute__((ext_vector_type(16))) __bf16 bf16x16;
typedef __attribute__((ext_vector_type(8)))  float  f32x8;

union FragU { uint4 u[2]; bf16x16 v; };

__device__ __forceinline__ uint16_t f2bf(float f) {
    union { float f; uint32_t u; } c; c.f = f;
    uint32_t r = (c.u + 0x7FFFu + ((c.u >> 16) & 1u)) >> 16;  // RNE
    return (uint16_t)r;
}

// Packed fp32x2 -> bf16x2 in one VALU op (v_cvt_pk_bf16_f32).
__device__ __forceinline__ uint32_t pk_bf16(float lo, float hi) {
    uint32_t v;
    asm("v_cvt_pk_bf16_f32 %0, %1, %2" : "=v"(v) : "v"(lo), "v"(hi));
    return v;
}

// LDS byte offset of a __shared__ object = low 32 bits of its flat address
// (ISA 10.2: LDS aperture addresses are truncated to addr[31:0]).
__device__ __forceinline__ uint32_t lds_off(const void* p) {
    return (uint32_t)(uintptr_t)p;
}

// CDNA5 async global->LDS copy, 16 bytes per lane, tracked by ASYNCcnt.
__device__ __forceinline__ void async_ld_b128(uint32_t dst_lds, const void* gsrc) {
    asm volatile("global_load_async_to_lds_b128 %0, %1, off"
                 :: "v"(dst_lds), "v"((unsigned long long)(uintptr_t)gsrc)
                 : "memory");
}
__device__ __forceinline__ void wait_async0() {
    asm volatile("s_wait_asynccnt 0x0" ::: "memory");
}

// ---------------------------------------------------------------------------
// Shared per-wave WMMA tile compute: 8 waves, block tile 128x128,
// wave tile 64x32 (4 M-subtiles x 2 N-subtiles), K-slice = 32 (bf16 WMMA).
// As: [128][32] bf16 row-major.  Bs: [128 n][32 k] bf16 (B pre-transposed).
// ---------------------------------------------------------------------------
__device__ __forceinline__ void wave_mma_step(const uint16_t* __restrict__ As,
                                              const uint16_t* __restrict__ Bs,
                                              int waveM, int waveN, int lane,
                                              f32x8 acc[4][2])
{
    const int l15  = lane & 15;
    const int half = lane >> 4;
    FragU a[4], b[2];
#pragma unroll
    for (int m = 0; m < 4; ++m) {
        const uint16_t* ap = &As[(waveM + m * 16 + l15) * 32];
        a[m].u[0] = *(const uint4*)(ap + half * 8);        // K 0-7 / 8-15
        a[m].u[1] = *(const uint4*)(ap + 16 + half * 8);   // K 16-23 / 24-31
    }
#pragma unroll
    for (int n = 0; n < 2; ++n) {
        const uint16_t* bp = &Bs[(waveN + n * 16 + l15) * 32 + half * 8];
        b[n].u[0] = *(const uint4*)(bp);
        b[n].u[1] = *(const uint4*)(bp + 16);
    }
#pragma unroll
    for (int m = 0; m < 4; ++m)
#pragma unroll
        for (int n = 0; n < 2; ++n)
            acc[m][n] = __builtin_amdgcn_wmma_f32_16x16x32_bf16(
                false, a[m].v, false, b[n].v, (short)0, acc[m][n], false, false);
}

// ---------------------------------------------------------------------------
// Kernel 1: QKV projection.  fp32 X [M=B*L, E] x fp32 W [E, N] -> bf16 [M, N]
// blockIdx.z in {0,1,2} selects Wq/Wk/Wv.  (VGPR staging path: needs fp32->bf16,
// packed via v_cvt_pk_bf16_f32.)
// ---------------------------------------------------------------------------
__global__ __launch_bounds__(256)
void k_qkv(const float* __restrict__ X,
           const float* __restrict__ Wq, const float* __restrict__ Wk,
           const float* __restrict__ Wv,
           uint16_t* __restrict__ Qo, uint16_t* __restrict__ Ko,
           uint16_t* __restrict__ Vo)
{
    const float*    W = (blockIdx.z == 0) ? Wq : (blockIdx.z == 1) ? Wk : Wv;
    uint16_t*       O = (blockIdx.z == 0) ? Qo : (blockIdx.z == 1) ? Ko : Vo;
    const int N = DK, Kd = E_DIM;
    const int m0 = blockIdx.y * 128, n0 = blockIdx.x * 128;

    __shared__ __align__(16) uint16_t As[128 * 32];
    __shared__ __align__(16) uint16_t Bs[128 * 32];   // [n][k]

    const int tid = threadIdx.x, lane = tid & 31, wid = tid >> 5;
    const int waveM = (wid & 1) * 64, waveN = (wid >> 1) * 32;

    f32x8 acc[4][2];
    const f32x8 z = {0.f, 0.f, 0.f, 0.f, 0.f, 0.f, 0.f, 0.f};
#pragma unroll
    for (int m = 0; m < 4; ++m) { acc[m][0] = z; acc[m][1] = z; }

    for (int k0 = 0; k0 < Kd; k0 += 32) {
        // Stage A tile (128x32 fp32 -> bf16), 16 floats / thread
#pragma unroll
        for (int i = 0; i < 4; ++i) {
            int idx = tid * 4 + i * 1024;                // element in [128][32]
            int r = idx >> 5, c = idx & 31;
            float4 f = *(const float4*)(X + (size_t)(m0 + r) * Kd + k0 + c);
            uint2 p;
            p.x = pk_bf16(f.x, f.y);
            p.y = pk_bf16(f.z, f.w);
            *(uint2*)&As[idx] = p;
        }
        // Stage B tile (W rows k0..k0+31, cols n0..n0+127) transposed -> Bs[n][k]
#pragma unroll
        for (int i = 0; i < 4; ++i) {
            int idx = tid * 4 + i * 1024;                // element in [32][128]
            int kk = idx >> 7, nn = idx & 127;
            float4 f = *(const float4*)(W + (size_t)(k0 + kk) * N + n0 + nn);
            uint32_t p01 = pk_bf16(f.x, f.y);
            uint32_t p23 = pk_bf16(f.z, f.w);
            Bs[(nn + 0) * 32 + kk] = (uint16_t)p01;
            Bs[(nn + 1) * 32 + kk] = (uint16_t)(p01 >> 16);
            Bs[(nn + 2) * 32 + kk] = (uint16_t)p23;
            Bs[(nn + 3) * 32 + kk] = (uint16_t)(p23 >> 16);
        }
        __syncthreads();

        if (k0 + 32 < Kd) {   // prefetch next K-slice (global_prefetch_b8)
            __builtin_prefetch(X + (size_t)(m0 + (tid & 127)) * Kd + k0 + 32, 0, 1);
            __builtin_prefetch(W + (size_t)(k0 + 32 + (tid & 31)) * N + n0 + ((tid >> 5) << 4), 0, 1);
        }

        wave_mma_step(As, Bs, waveM, waveN, lane, acc);
        __syncthreads();
    }

    // Store bf16 result (C/D layout: vgpr r -> row (lane>=16 ? r+8 : r))
#pragma unroll
    for (int m = 0; m < 4; ++m)
#pragma unroll
        for (int n = 0; n < 2; ++n) {
            int col   = n0 + waveN + n * 16 + (lane & 15);
            int rbase = m0 + waveM + m * 16 + (lane >> 4) * 8;
#pragma unroll
            for (int r = 0; r < 8; r += 2) {
                uint32_t pr = pk_bf16(acc[m][n][r], acc[m][n][r + 1]);
                O[(size_t)(rbase + r) * N + col]     = (uint16_t)pr;
                O[(size_t)(rbase + r + 1) * N + col] = (uint16_t)(pr >> 16);
            }
        }
}

// ---------------------------------------------------------------------------
// Kernel 2: scores S = Q . K^T (per batch), fp32 out. Skip fully-masked tiles.
// Both A and B staging are raw bf16 copies -> double-buffered ASYNC loads to
// LDS (global_load_async_to_lds_b128, ASYNCcnt) overlapped with the WMMAs.
// ---------------------------------------------------------------------------
__global__ __launch_bounds__(256)
void k_scores(const uint16_t* __restrict__ Q, const uint16_t* __restrict__ Kc,
              float* __restrict__ S)
{
    const int kt = blockIdx.x, qt = blockIdx.y, b = blockIdx.z;
    if (kt > qt) return;                       // entirely above causal diagonal
    const uint16_t* Qb = Q  + (size_t)b * SEQ * DK;
    const uint16_t* Kb = Kc + (size_t)b * SEQ * DK;
    float*          Sb = S  + (size_t)b * SEQ * SEQ;
    const int m0 = qt * 128, n0 = kt * 128;

    __shared__ __align__(16) uint16_t As[2][128 * 32];
    __shared__ __align__(16) uint16_t Bs[2][128 * 32];

    const int tid = threadIdx.x, lane = tid & 31, wid = tid >> 5;
    const int waveM = (wid & 1) * 64, waveN = (wid >> 1) * 32;

    f32x8 acc[4][2];
    const f32x8 z = {0.f, 0.f, 0.f, 0.f, 0.f, 0.f, 0.f, 0.f};
#pragma unroll
    for (int m = 0; m < 4; ++m) { acc[m][0] = z; acc[m][1] = z; }

    // Per-thread staging slots: 2 x b128 (8 bf16) for each of A and B.
    const int idx0 = tid * 8, idx1 = tid * 8 + 2048;
    const int r0 = idx0 >> 5, c0 = idx0 & 31;
    const int r1 = idx1 >> 5, c1 = idx1 & 31;

    // Preload K-slice 0 into buffer 0
    async_ld_b128(lds_off(&As[0][idx0]), Qb + (size_t)(m0 + r0) * DK + c0);
    async_ld_b128(lds_off(&As[0][idx1]), Qb + (size_t)(m0 + r1) * DK + c1);
    async_ld_b128(lds_off(&Bs[0][idx0]), Kb + (size_t)(n0 + r0) * DK + c0);
    async_ld_b128(lds_off(&Bs[0][idx1]), Kb + (size_t)(n0 + r1) * DK + c1);
    wait_async0();
    __syncthreads();

    int cur = 0;
    for (int k0 = 0; k0 < DK; k0 += 32) {
        const int nxt = cur ^ 1;
        if (k0 + 32 < DK) {   // kick off next slice while we compute this one
            const int kn = k0 + 32;
            async_ld_b128(lds_off(&As[nxt][idx0]), Qb + (size_t)(m0 + r0) * DK + kn + c0);
            async_ld_b128(lds_off(&As[nxt][idx1]), Qb + (size_t)(m0 + r1) * DK + kn + c1);
            async_ld_b128(lds_off(&Bs[nxt][idx0]), Kb + (size_t)(n0 + r0) * DK + kn + c0);
            async_ld_b128(lds_off(&Bs[nxt][idx1]), Kb + (size_t)(n0 + r1) * DK + kn + c1);
        }
        wave_mma_step(As[cur], Bs[cur], waveM, waveN, lane, acc);
        wait_async0();         // next buffer's data has landed in LDS
        __syncthreads();       // everyone done reading cur + sees nxt
        cur = nxt;
    }

#pragma unroll
    for (int m = 0; m < 4; ++m)
#pragma unroll
        for (int n = 0; n < 2; ++n) {
            int col   = n0 + waveN + n * 16 + (lane & 15);
            int rbase = m0 + waveM + m * 16 + (lane >> 4) * 8;
#pragma unroll
            for (int r = 0; r < 8; ++r)
                Sb[(size_t)(rbase + r) * SEQ + col] = acc[m][n][r];
        }
}

// ---------------------------------------------------------------------------
// Kernel 3: causal row softmax. One 256-thread block per (b, q).
// Writes bf16 probabilities; zero-fills to the 128-aligned tile edge so k_pv
// can consume tile-granular K ranges.
// ---------------------------------------------------------------------------
__global__ __launch_bounds__(256)
void k_softmax(const float* __restrict__ S, uint16_t* __restrict__ P)
{
    const int q = blockIdx.x, b = blockIdx.y;
    const float* row  = S + ((size_t)b * SEQ + q) * SEQ;
    uint16_t*    prow = P + ((size_t)b * SEQ + q) * SEQ;
    const int n   = q + 1;
    const int lim = ((q >> 7) + 1) << 7;       // tile-aligned end
    const int tid = threadIdx.x;

    __shared__ float red[256];

    float m = -3.4e38f;
    for (int j = tid; j < n; j += 256) m = fmaxf(m, row[j]);
    red[tid] = m; __syncthreads();
    for (int s = 128; s > 0; s >>= 1) {
        if (tid < s) red[tid] = fmaxf(red[tid], red[tid + s]);
        __syncthreads();
    }
    m = red[0]; __syncthreads();

    float sum = 0.f;
    for (int j = tid; j < n; j += 256) sum += __expf((row[j] - m) * SM_SCALE);
    red[tid] = sum; __syncthreads();
    for (int s = 128; s > 0; s >>= 1) {
        if (tid < s) red[tid] += red[tid + s];
        __syncthreads();
    }
    const float inv = 1.0f / red[0];

    for (int j = tid; j < lim; j += 256)
        prow[j] = (j < n) ? f2bf(__expf((row[j] - m) * SM_SCALE) * inv)
                          : (uint16_t)0;
}

// ---------------------------------------------------------------------------
// Kernel 4: O = P . V (per batch), fp32 out, K-loop truncated causally.
// A (P rows) staged with ASYNC LDS loads; V tile transpose-staged via VGPRs.
// ---------------------------------------------------------------------------
__global__ __launch_bounds__(256)
void k_pv(const uint16_t* __restrict__ P, const uint16_t* __restrict__ V,
          float* __restrict__ Out)
{
    const int qt = blockIdx.y, b = blockIdx.z;
    const int n0 = blockIdx.x * 128, m0 = qt * 128;
    const int kmax = (qt + 1) * 128;
    const uint16_t* Pb = P + (size_t)b * SEQ * SEQ;
    const uint16_t* Vb = V + (size_t)b * SEQ * DV;
    float*          Ob = Out + (size_t)b * SEQ * DV;

    __shared__ __align__(16) uint16_t As[128 * 32];
    __shared__ __align__(16) uint16_t Bs[128 * 32];   // [n][k]

    const int tid = threadIdx.x, lane = tid & 31, wid = tid >> 5;
    const int waveM = (wid & 1) * 64, waveN = (wid >> 1) * 32;

    f32x8 acc[4][2];
    const f32x8 z = {0.f, 0.f, 0.f, 0.f, 0.f, 0.f, 0.f, 0.f};
#pragma unroll
    for (int m = 0; m < 4; ++m) { acc[m][0] = z; acc[m][1] = z; }

    const int idx0 = tid * 8, idx1 = tid * 8 + 2048;
    const int r0 = idx0 >> 5, c0 = idx0 & 31;
    const int r1 = idx1 >> 5, c1 = idx1 & 31;

    for (int k0 = 0; k0 < kmax; k0 += 32) {
        // A: P rows, raw bf16 copy -> async LDS loads
        async_ld_b128(lds_off(&As[idx0]), Pb + (size_t)(m0 + r0) * SEQ + k0 + c0);
        async_ld_b128(lds_off(&As[idx1]), Pb + (size_t)(m0 + r1) * SEQ + k0 + c1);

        // B: V rows k0..k0+31, cols n0..n0+127 -> transpose into Bs[n][k]
#pragma unroll
        for (int i = 0; i < 2; ++i) {
            int idx = tid * 8 + i * 2048;      // element in [32][128]
            int kk = idx >> 7, nn = idx & 127;
            uint4 raw = *(const uint4*)(Vb + (size_t)(k0 + kk) * DV + n0 + nn);
            const uint16_t* h = (const uint16_t*)&raw;
#pragma unroll
            for (int j = 0; j < 8; ++j)
                Bs[(nn + j) * 32 + kk] = h[j];
        }

        if (k0 + 32 < kmax) {
            __builtin_prefetch(Vb + (size_t)(k0 + 32 + (tid & 31)) * DV + n0 + ((tid >> 5) << 4), 0, 1);
        }

        wait_async0();
        __syncthreads();
        wave_mma_step(As, Bs, waveM, waveN, lane, acc);
        __syncthreads();
    }

#pragma unroll
    for (int m = 0; m < 4; ++m)
#pragma unroll
        for (int n = 0; n < 2; ++n) {
            int col   = n0 + waveN + n * 16 + (lane & 15);
            int rbase = m0 + waveM + m * 16 + (lane >> 4) * 8;
#pragma unroll
            for (int r = 0; r < 8; ++r)
                Ob[(size_t)(rbase + r) * DV + col] = acc[m][n][r];
        }
}

// ---------------------------------------------------------------------------
// Host entry
// ---------------------------------------------------------------------------
extern "C" void kernel_launch(void* const* d_in, const int* in_sizes, int n_in,
                              void* d_out, int out_size, void* d_ws, size_t ws_size,
                              hipStream_t stream)
{
    const float* x  = (const float*)d_in[0];
    const float* Wq = (const float*)d_in[1];
    const float* Wk = (const float*)d_in[2];
    const float* Wv = (const float*)d_in[3];
    float* out = (float*)d_out;

    // Workspace layout: Q,K,V bf16 (3 x 64 MB), S fp32 (256 MB), P bf16 (128 MB)
    uint16_t* Qb = (uint16_t*)d_ws;
    uint16_t* Kb = Qb + (size_t)BATCH * SEQ * DK;
    uint16_t* Vb = Kb + (size_t)BATCH * SEQ * DK;
    float*    S  = (float*)(Vb + (size_t)BATCH * SEQ * DV);
    uint16_t* Pp = (uint16_t*)(S + (size_t)BATCH * SEQ * SEQ);

    dim3 blk(256);
    k_qkv   <<<dim3(DK / 128, (BATCH * SEQ) / 128, 3), blk, 0, stream>>>(x, Wq, Wk, Wv, Qb, Kb, Vb);
    k_scores<<<dim3(SEQ / 128, SEQ / 128, BATCH),      blk, 0, stream>>>(Qb, Kb, S);
    k_softmax<<<dim3(SEQ, BATCH),                      blk, 0, stream>>>(S, Pp);
    k_pv    <<<dim3(DV / 128, SEQ / 128, BATCH),       blk, 0, stream>>>(Pp, Vb, out);
}
